// Model_10668698763867
// MI455X (gfx1250) — matrix operand order repaired and checked by reading, not compiled, not run on hardware
//
#include <hip/hip_runtime.h>
#include <hip/hip_bf16.h>
#include <stdint.h>

typedef __attribute__((ext_vector_type(16))) _Float16 v16h;
typedef __attribute__((ext_vector_type(8)))  float    v8f;

#define T_STEPS 1200
#define NBATCH  32
#define HDIM    384
#define GDIM    1536   // 4*H
#define NSCORE  1280
#define MROWS   (T_STEPS * NBATCH)   // 38400

__device__ __forceinline__ float sigf(float x) { return 1.0f / (1.0f + expf(-x)); }
__device__ __forceinline__ float siluf(float x) { return x * sigf(x); }

// ---------------- WMMA tile loaders (contiguous k order: k = 16*(lane/16) + e, same for A and B) ----------------
// A: 16x32 f16 from row-major src (lda elements). lane<16 holds row lane with
// K {kbase..kbase+15}, kbase = 16*(lane>=16).
__device__ __forceinline__ v16h load_a_rm(const _Float16* A, int lda, int row0, int k0) {
    int lane  = threadIdx.x & 31;
    int m     = lane & 15;
    int kbase = ((lane >> 4) & 1) << 4;
    const _Float16* p = A + (size_t)(row0 + m) * lda + k0 + kbase;
    union { v16h v; _Float16 h[16]; } r;
#pragma unroll
    for (int e = 0; e < 8; ++e)  r.h[e] = p[e];
#pragma unroll
    for (int e = 8; e < 16; ++e) r.h[e] = p[e];
    return r.v;
}

// B tile (32x16, K x N) where B[k][n] = W[col0+n][k0+k], W row-major (ldw).
// Per lane: n = lane&15, k = 16*(lane>=16) + e  -> 16 contiguous halves.
__device__ __forceinline__ v16h load_b_wt(const _Float16* W, int ldw, int col0, int k0) {
    int lane = threadIdx.x & 31;
    int n    = lane & 15;
    int kb   = (lane >> 4) << 4;
    const _Float16* p = W + (size_t)(col0 + n) * ldw + k0 + kb;
    union { v16h v; _Float16 h[16]; } r;
#pragma unroll
    for (int e = 0; e < 16; ++e) r.h[e] = p[e];
    return r.v;
}

// ---------------- conv1 + conv2 (fused, SiLU) ----------------
__global__ __launch_bounds__(256) void conv12_kernel(
    const float* __restrict__ x, const float* __restrict__ w1, const float* __restrict__ b1,
    const float* __restrict__ w2, const float* __restrict__ b2, float* __restrict__ c2out)
{
    int id = blockIdx.x * 256 + threadIdx.x;
    if (id >= NBATCH * 6000) return;
    int b = id / 6000, t = id % 6000;
    float xv[9];
#pragma unroll
    for (int k = 0; k < 9; ++k) {
        int tt = t - 4 + k;
        xv[k] = (tt >= 0 && tt < 6000) ? x[(size_t)b * 6000 + tt] : 0.f;
    }
    float c1v[20];
#pragma unroll
    for (int c = 0; c < 4; ++c) {
#pragma unroll
        for (int p = 0; p < 5; ++p) {
            float a = b1[c];
#pragma unroll
            for (int k = 0; k < 5; ++k) a += xv[p + k] * w1[c * 5 + k];
            int pos = t - 2 + p;
            c1v[c * 5 + p] = (pos >= 0 && pos < 6000) ? siluf(a) : 0.f;
        }
    }
#pragma unroll
    for (int c2 = 0; c2 < 16; ++c2) {
        float a = b2[c2];
#pragma unroll
        for (int c = 0; c < 4; ++c)
#pragma unroll
            for (int k = 0; k < 5; ++k)
                a += c1v[c * 5 + k] * w2[c2 * 20 + c * 5 + k];
        c2out[(size_t)b * 96000 + (size_t)c2 * 6000 + t] = siluf(a);
    }
}

// ---------------- conv3 (stride 5, k=19, SiLU) -> feat f16 (T,N,H) ----------------
__global__ __launch_bounds__(128) void conv3_kernel(
    const float* __restrict__ c2, const float* __restrict__ w3, const float* __restrict__ b3,
    _Float16* __restrict__ feat)
{
    __shared__ float win[304 * 32];   // [c*19+k][n]
    int tb = blockIdx.x;              // output time
    int h  = blockIdx.y * 128 + threadIdx.x;
    for (int i = threadIdx.x; i < 304 * 32; i += 128) {
        int ck = i >> 5, n = i & 31;
        int c = ck / 19, k = ck % 19;
        int tt = 5 * tb - 9 + k;
        win[i] = (tt >= 0 && tt < 6000) ? c2[(size_t)n * 96000 + (size_t)c * 6000 + tt] : 0.f;
    }
    __syncthreads();
    float acc[32];
#pragma unroll
    for (int n = 0; n < 32; ++n) acc[n] = 0.f;
    for (int ck = 0; ck < 304; ++ck) {
        float wv = w3[(size_t)h * 304 + ck];
        const float* wp = &win[ck * 32];
#pragma unroll
        for (int n = 0; n < 32; ++n) acc[n] += wp[n] * wv;
    }
    float bv = b3[h];
#pragma unroll
    for (int n = 0; n < 32; ++n)
        feat[((size_t)tb * 32 + n) * HDIM + h] = (_Float16)siluf(acc[n] + bv);
}

// ---------------- f32 -> f16 weight convert ----------------
__global__ void cvt_f16_kernel(const float* __restrict__ s, _Float16* __restrict__ d, int n) {
    int i = blockIdx.x * 256 + threadIdx.x;
    if (i < n) d[i] = (_Float16)s[i];
}

__global__ void zero_u32_kernel(unsigned* p) { if (threadIdx.x == 0 && blockIdx.x == 0) *p = 0u; }

// ---------------- big GEMM: D = A(38400xK384) @ W^T + bias ----------------
// mode 0: store f16 to outh (LSTM x@wih precompute).  mode 1: 5*tanh, f32 to outf.
__global__ __launch_bounds__(256) void gemm_wt_kernel(
    const _Float16* __restrict__ A, const _Float16* __restrict__ W,
    const float* __restrict__ bias, _Float16* __restrict__ outh,
    float* __restrict__ outf, int Nout, int mode)
{
    int w    = threadIdx.x >> 5;
    int lane = threadIdx.x & 31;
    int row0 = blockIdx.x * 64 + (w >> 2) * 32;
    int col0 = blockIdx.y * 128 + (w & 3) * 32;
    v8f c00 = {}, c01 = {}, c10 = {}, c11 = {};
    for (int kc = 0; kc < HDIM; kc += 32) {
        if (kc + 32 < HDIM)
            __builtin_prefetch(A + (size_t)row0 * HDIM + kc + 32, 0, 3);
        v16h a0 = load_a_rm(A, HDIM, row0, kc);
        v16h a1 = load_a_rm(A, HDIM, row0 + 16, kc);
        v16h b0 = load_b_wt(W, HDIM, col0, kc);
        v16h b1 = load_b_wt(W, HDIM, col0 + 16, kc);
        c00 = __builtin_amdgcn_wmma_f32_16x16x32_f16(false, a0, false, b0, (short)0, c00, false, false);
        c01 = __builtin_amdgcn_wmma_f32_16x16x32_f16(false, a0, false, b1, (short)0, c01, false, false);
        c10 = __builtin_amdgcn_wmma_f32_16x16x32_f16(false, a1, false, b0, (short)0, c10, false, false);
        c11 = __builtin_amdgcn_wmma_f32_16x16x32_f16(false, a1, false, b1, (short)0, c11, false, false);
    }
    int n   = lane & 15;
    int mhi = (lane >> 4) << 3;
    auto emit = [&](v8f cc, int rbase, int cbase) {
        union { v8f v; float f[8]; } u; u.v = cc;
        int col  = col0 + cbase + n;
        float bv = bias[col];
#pragma unroll
        for (int v2 = 0; v2 < 8; ++v2) {
            int r = row0 + rbase + mhi + v2;
            float val = u.f[v2] + bv;
            if (mode == 0) outh[(size_t)r * Nout + col] = (_Float16)val;
            else           outf[(size_t)r * Nout + col] = 5.0f * tanhf(val);
        }
    };
    emit(c00, 0, 0); emit(c01, 0, 16); emit(c10, 16, 0); emit(c11, 16, 16);
}

// ---------------- persistent LSTM recurrence ----------------
// 12 WGs x 256 thr. WG owns 32 hidden cols (all 4 gates); whh slice in LDS.
// Grid barrier each step via device atomics (all 12 WGs co-resident).
__global__ __launch_bounds__(256) void lstm_rec_kernel(
    const _Float16* __restrict__ xw, const _Float16* __restrict__ whh,
    _Float16* __restrict__ y, int reverse, unsigned* __restrict__ bar)
{
    extern __shared__ char smem[];
    _Float16* Ws   = (_Float16*)smem;                             // [128][384]
    _Float16* hs   = (_Float16*)(smem + 98304);                   // [32][384]
    float*    gts  = (float*)(smem + 98304 + 24576);              // [32][128]
    float*    cst  = (float*)(smem + 98304 + 24576 + 16384);      // [32][32]
    int tid = threadIdx.x;
    int j0  = blockIdx.x * 32;
    // cache whh slice: local col jj = g*32+jc <-> global row g*384+j0+jc
    for (int i = tid; i < 128 * HDIM; i += 256) {
        int jj = i / HDIM, k = i % HDIM;
        int g = jj >> 5, jc = jj & 31;
        Ws[i] = whh[(size_t)(g * HDIM + j0 + jc) * HDIM + k];
    }
    for (int i = tid; i < 1024; i += 256) cst[i] = 0.f;
    __syncthreads();
    int wav = tid >> 5, lane = tid & 31;
    unsigned target = 0;
    for (int step = 0; step < T_STEPS; ++step) {
        int t = reverse ? (T_STEPS - 1 - step) : step;
        if (step == 0) {
            for (int i = tid; i < NBATCH * HDIM; i += 256) hs[i] = (_Float16)0.f;
        } else {
            int tp = reverse ? (t + 1) : (t - 1);
            const uint32_t* src = (const uint32_t*)(y + (size_t)tp * NBATCH * HDIM);
            uint32_t* dst = (uint32_t*)hs;
            for (int i = tid; i < NBATCH * HDIM / 2; i += 256) dst[i] = src[i];
        }
        __syncthreads();
        // gates[0:32, wav*16 : wav*16+16] = hs(32x384) @ Ws^T slice
        v8f c0 = {}, c1 = {};
        int lcol = wav * 16;
        for (int kc = 0; kc < HDIM; kc += 32) {
            v16h b  = load_b_wt(Ws, HDIM, lcol, kc);
            v16h a0 = load_a_rm(hs, HDIM, 0, kc);
            v16h a1 = load_a_rm(hs, HDIM, 16, kc);
            c0 = __builtin_amdgcn_wmma_f32_16x16x32_f16(false, a0, false, b, (short)0, c0, false, false);
            c1 = __builtin_amdgcn_wmma_f32_16x16x32_f16(false, a1, false, b, (short)0, c1, false, false);
        }
        int n = lane & 15, mhi = (lane >> 4) << 3;
        int lc = lcol + n;
        int g = lc >> 5, jc = lc & 31;
        int gcol = g * HDIM + j0 + jc;
        const _Float16* xwt = xw + (size_t)t * NBATCH * GDIM + gcol;
        union { v8f v; float f[8]; } u0, u1; u0.v = c0; u1.v = c1;
#pragma unroll
        for (int v2 = 0; v2 < 8; ++v2) {
            int r0 = mhi + v2;
            gts[r0 * 128 + lc] = u0.f[v2] + (float)xwt[(size_t)r0 * GDIM];
            int r1 = 16 + mhi + v2;
            gts[r1 * 128 + lc] = u1.f[v2] + (float)xwt[(size_t)r1 * GDIM];
        }
        __syncthreads();
        // cell update for owned 32 cols
        _Float16* yt = y + (size_t)t * NBATCH * HDIM;
#pragma unroll
        for (int q = 0; q < 4; ++q) {
            int id = tid + q * 256;
            int nn = id >> 5, j = id & 31;
            float gi = gts[nn * 128 + j];
            float gf = gts[nn * 128 + 32 + j];
            float gg = gts[nn * 128 + 64 + j];
            float go = gts[nn * 128 + 96 + j];
            float cv = sigf(gf) * cst[nn * 32 + j] + sigf(gi) * tanhf(gg);
            cst[nn * 32 + j] = cv;
            yt[(size_t)nn * HDIM + j0 + j] = (_Float16)(sigf(go) * tanhf(cv));
        }
        __syncthreads();
        // device-wide barrier (12 WGs)
        target += gridDim.x;
        if (tid == 0) {
            __threadfence();
            atomicAdd(bar, 1u);
            while (atomicAdd(bar, 0u) < target) __builtin_amdgcn_s_sleep(1);
            __threadfence();
        }
        __syncthreads();
    }
}

// ---------------- CRF forward scan (one block per batch element) ----------------
__global__ __launch_bounds__(256) void crf_kernel(
    const float* __restrict__ scores, const int* __restrict__ idx, float* __restrict__ logz)
{
    __shared__ float al[2][256];
    __shared__ int   sidx[1280];
    __shared__ float red[256];
    int s = threadIdx.x, n = blockIdx.x;
    for (int i = s; i < 1280; i += 256) sidx[i] = idx[i];
    al[0][s] = 0.f;
    __syncthreads();
    int cur = 0;
    for (int t = 0; t < T_STEPS; ++t) {
        const float* Mt = scores + ((size_t)t * NBATCH + n) * NSCORE + s * 5;
        float v0 = al[cur][sidx[s * 5 + 0]] + Mt[0];
        float v1 = al[cur][sidx[s * 5 + 1]] + Mt[1];
        float v2 = al[cur][sidx[s * 5 + 2]] + Mt[2];
        float v3 = al[cur][sidx[s * 5 + 3]] + Mt[3];
        float v4 = al[cur][sidx[s * 5 + 4]] + Mt[4];
        float mx = fmaxf(fmaxf(fmaxf(v0, v1), fmaxf(v2, v3)), v4);
        float sm = expf(v0 - mx) + expf(v1 - mx) + expf(v2 - mx) + expf(v3 - mx) + expf(v4 - mx);
        al[cur ^ 1][s] = mx + logf(sm);
        cur ^= 1;
        __syncthreads();
    }
    float a = al[cur][s];
    red[s] = a; __syncthreads();
    for (int off = 128; off > 0; off >>= 1) { if (s < off) red[s] = fmaxf(red[s], red[s + off]); __syncthreads(); }
    float mx = red[0]; __syncthreads();
    red[s] = expf(a - mx); __syncthreads();
    for (int off = 128; off > 0; off >>= 1) { if (s < off) red[s] += red[s + off]; __syncthreads(); }
    if (s == 0) logz[n] = mx + logf(red[0]);
}

__global__ void sub_logz_kernel(float* __restrict__ out, const float* __restrict__ logz, size_t total) {
    size_t i = (size_t)blockIdx.x * 256 + threadIdx.x;
    if (i < total) {
        int n = (int)((i / NSCORE) & 31);
        out[i] -= logz[n] * (1.0f / (float)T_STEPS);
    }
}

extern "C" void kernel_launch(void* const* d_in, const int* in_sizes, int n_in,
                              void* d_out, int out_size, void* d_ws, size_t ws_size,
                              hipStream_t stream) {
    (void)in_sizes; (void)n_in; (void)out_size; (void)ws_size;
    const float* x   = (const float*)d_in[0];
    const float* w1  = (const float*)d_in[1];
    const float* b1  = (const float*)d_in[2];
    const float* w2  = (const float*)d_in[3];
    const float* b2  = (const float*)d_in[4];
    const float* w3  = (const float*)d_in[5];
    const float* b3  = (const float*)d_in[6];
    const float* lw  = (const float*)d_in[7];
    const float* lb  = (const float*)d_in[8];
    const int*   idx = (const int*)d_in[24];
    float* out = (float*)d_out;

    char* ws = (char*)d_ws;
    size_t o = 0;
    _Float16* bufA   = (_Float16*)(ws + o); o += (size_t)MROWS * HDIM * 2;     // 29.5 MB
    _Float16* bufB   = (_Float16*)(ws + o); o += (size_t)MROWS * HDIM * 2;     // 29.5 MB
    _Float16* xw     = (_Float16*)(ws + o); o += (size_t)MROWS * GDIM * 2;     // 118 MB
    float*    c2out  = (float*)(ws + o);    o += (size_t)NBATCH * 16 * 6000 * 4;
    _Float16* wih_h  = (_Float16*)(ws + o); o += (size_t)GDIM * HDIM * 2;
    _Float16* whh_h  = (_Float16*)(ws + o); o += (size_t)GDIM * HDIM * 2;
    _Float16* linw_h = (_Float16*)(ws + o); o += (size_t)NSCORE * HDIM * 2;
    float*    logz   = (float*)(ws + o);    o += 256;
    unsigned* bar    = (unsigned*)(ws + o); o += 256;

    // convs -> feat f16 in bufA
    conv12_kernel<<<(NBATCH * 6000 + 255) / 256, 256, 0, stream>>>(x, w1, b1, w2, b2, c2out);
    conv3_kernel<<<dim3(T_STEPS, HDIM / 128), 128, 0, stream>>>(c2out, w3, b3, bufA);
    cvt_f16_kernel<<<(NSCORE * HDIM + 255) / 256, 256, 0, stream>>>(lw, linw_h, NSCORE * HDIM);

    const int rev[5] = {1, 0, 1, 0, 1};
    for (int L = 0; L < 5; ++L) {
        const float* wih = (const float*)d_in[9 + 3 * L];
        const float* whh = (const float*)d_in[10 + 3 * L];
        const float* lbb = (const float*)d_in[11 + 3 * L];
        _Float16* ain  = (L & 1) ? bufB : bufA;
        _Float16* yout = (L & 1) ? bufA : bufB;
        cvt_f16_kernel<<<(GDIM * HDIM + 255) / 256, 256, 0, stream>>>(wih, wih_h, GDIM * HDIM);
        cvt_f16_kernel<<<(GDIM * HDIM + 255) / 256, 256, 0, stream>>>(whh, whh_h, GDIM * HDIM);
        gemm_wt_kernel<<<dim3(MROWS / 64, GDIM / 128), 256, 0, stream>>>(
            ain, wih_h, lbb, xw, nullptr, GDIM, 0);
        zero_u32_kernel<<<1, 64, 0, stream>>>(bar);
        lstm_rec_kernel<<<12, 256, 98304 + 24576 + 16384 + 4096, stream>>>(
            xw, whh_h, yout, rev[L], bar);
    }
    // final linear + 5*tanh -> scores into d_out (f32)
    gemm_wt_kernel<<<dim3(MROWS / 64, NSCORE / 128), 256, 0, stream>>>(
        bufB, linw_h, lb, nullptr, out, NSCORE, 1);
    // CRF logZ, then subtract logz/T
    crf_kernel<<<NBATCH, 256, 0, stream>>>(out, idx, logz);
    size_t total = (size_t)MROWS * NSCORE;
    sub_logz_kernel<<<(unsigned)((total + 255) / 256), 256, 0, stream>>>(out, logz, total);
}
